// GQMultiheadAttention_88682484728632
// MI455X (gfx1250) — matrix operand order repaired and checked by reading, not compiled, not run on hardware
//
#include <hip/hip_runtime.h>
#include <hip/hip_bf16.h>
#include <cstdint>

// ---------------------------------------------------------------------------
// GQA multi-head attention for MI455X (gfx1250, wave32, WMMA).
//   B=2, S=2048, D_MODEL=2048, H=16, G=4 (KV heads), HD=128.
// Pipeline:
//   1) gemm_f16_wmma : Q/K/V projections (f32 in, f32 out) via v_wmma_f32_16x16x32_f16,
//                      register-pipelined global->LDS staging
//   2) rope_pack     : RoPE on Q/K, convert to f16, transpose to [B,H,S,HD]/[B,G,S,HD]
//   3) attn_wmma     : flash attention, 16 WMMAs per 32-key tile per wave
//   4) gemm_f16_wmma : output projection -> d_out (f32)
// ---------------------------------------------------------------------------

typedef __attribute__((ext_vector_type(16))) _Float16 v16h;
typedef __attribute__((ext_vector_type(8)))  float    v8f;

union Frag16 { uint32_t u[8]; v16h v; };

__device__ __forceinline__ uint32_t pack_f16(float a, float b) {
  union { _Float16 h[2]; uint32_t u; } r;
  r.h[0] = (_Float16)a; r.h[1] = (_Float16)b;
  return r.u;
}
__device__ __forceinline__ uint16_t f16_bits(float a) {
  union { _Float16 h; uint16_t u; } r;
  r.h = (_Float16)a;
  return r.u;
}

#define WMMA_F16(A, B, C) \
  __builtin_amdgcn_wmma_f32_16x16x32_f16(false, (A), false, (B), (short)0, (C), false, false)

// ---------------------------------------------------------------------------
// Kernel 1/4: C[M,N] = A[M,K] (f32) @ W[K,N] (f32) + bias, f32 out.
// Block = 256 threads (8 waves), tile 128x128, K-step 32.
// Wave (wm,wn) in 2x4 grid owns a 64x32 subtile = 4x2 WMMA accumulators.
// Global loads for tile t+1 are issued right after the barrier so they overlap
// the 8 WMMAs (and their LDS fragment loads) of tile t.
// ---------------------------------------------------------------------------
__global__ __launch_bounds__(256) void gemm_f16_wmma(
    const float* __restrict__ A, const float* __restrict__ W,
    const float* __restrict__ bias, float* __restrict__ C,
    int M, int N, int K) {
  __shared__ uint32_t Al[128][17];  // [row][k-pair], f16x2 packed, padded
  __shared__ uint32_t Bl[128][17];  // [col][k-pair], f16x2 packed, padded

  const int tid  = threadIdx.x;
  const int lane = tid & 31;
  const int wave = tid >> 5;
  const int wm   = wave & 1;   // 0..1 -> 64-row half
  const int wn   = wave >> 1;  // 0..3 -> 32-col quarter
  const int l16  = lane & 15;
  const int hl   = lane >> 4;  // half-wave select
  const int m0   = blockIdx.y * 128;
  const int n0   = blockIdx.x * 128;

  v8f acc[4][2];
#pragma unroll
  for (int mt = 0; mt < 4; ++mt)
#pragma unroll
    for (int nt = 0; nt < 2; ++nt)
      acc[mt][nt] = (v8f){0.f, 0.f, 0.f, 0.f, 0.f, 0.f, 0.f, 0.f};

  const int arow = tid >> 1;          // 0..127
  const int apg  = (tid & 1) * 8;     // pair group 0 or 8
  const int bn   = tid & 127;         // 0..127
  const int bpg  = (tid >> 7) * 8;    // pair group 0 or 8

  // Raw f32 staging registers for the next tile (pipelined).
  float aF[16], bF[16];

  auto load_tiles = [&](int k0) {
    const float* Ap = A + (size_t)(m0 + arow) * K + k0 + apg * 2;
#pragma unroll
    for (int j = 0; j < 16; ++j) aF[j] = Ap[j];
#pragma unroll
    for (int j = 0; j < 8; ++j) {
      const int p = bpg + j;
      const float* Wp = W + (size_t)(k0 + 2 * p) * N + n0 + bn;
      bF[2 * j]     = Wp[0];
      bF[2 * j + 1] = Wp[N];
    }
  };

  load_tiles(0);
  const int nk = K / 32;
  for (int kt = 0; kt < nk; ++kt) {
    __syncthreads();  // previous iteration's readers done with LDS
    // Pack staged f32 -> f16x2 and store to LDS.
#pragma unroll
    for (int j = 0; j < 8; ++j)
      Al[arow][apg + j] = pack_f16(aF[2 * j], aF[2 * j + 1]);
#pragma unroll
    for (int j = 0; j < 8; ++j)
      Bl[bn][bpg + j] = pack_f16(bF[2 * j], bF[2 * j + 1]);
    __syncthreads();
    // Issue next tile's global loads now; they complete during the WMMAs.
    if (kt + 1 < nk) load_tiles((kt + 1) * 32);

    // A fragments: lane<16 -> pairs {0..3, 8..11}; lane>=16 -> {4..7, 12..15}.
    Frag16 Af[4];
#pragma unroll
    for (int mt = 0; mt < 4; ++mt) {
      const int row  = wm * 64 + mt * 16 + l16;
      const int base = hl ? 4 : 0;
#pragma unroll
      for (int i = 0; i < 4; ++i) {
        Af[mt].u[i]     = Al[row][base + i];
        Af[mt].u[4 + i] = Al[row][base + 8 + i];
      }
    }
    // B fragments: lane<16 -> pairs {0..3, 8..11}; lane>=16 -> {4..7, 12..15}.
    Frag16 Bf[2];
#pragma unroll
    for (int nt = 0; nt < 2; ++nt) {
      const int col  = wn * 32 + nt * 16 + l16;
      const int base = hl ? 4 : 0;
#pragma unroll
      for (int i = 0; i < 4; ++i) {
        Bf[nt].u[i]     = Bl[col][base + i];
        Bf[nt].u[4 + i] = Bl[col][base + 8 + i];
      }
    }
#pragma unroll
    for (int mt = 0; mt < 4; ++mt)
#pragma unroll
      for (int nt = 0; nt < 2; ++nt)
        acc[mt][nt] = WMMA_F16(Af[mt].v, Bf[nt].v, acc[mt][nt]);
  }

  // Epilogue: C layout — lane l holds col n=l%16, rows r (+8 for upper half).
#pragma unroll
  for (int mt = 0; mt < 4; ++mt) {
#pragma unroll
    for (int nt = 0; nt < 2; ++nt) {
      const int col = n0 + wn * 32 + nt * 16 + l16;
      const float bv = bias[col];
#pragma unroll
      for (int r = 0; r < 8; ++r) {
        const int row = m0 + wm * 64 + mt * 16 + hl * 8 + r;
        C[(size_t)row * N + col] = acc[mt][nt][r] + bv;
      }
    }
  }
}

// ---------------------------------------------------------------------------
// Kernel 2: RoPE on Q/K, f32 -> f16, transpose to attention layouts.
// One thread per (even,odd) column pair. 1536 pairs per (b,s) row:
//   [0,1024): Q pairs, [1024,1280): K pairs, [1280,1536): V pairs.
// ---------------------------------------------------------------------------
__global__ __launch_bounds__(256) void rope_pack(
    const float* __restrict__ Qf, const float* __restrict__ Kf,
    const float* __restrict__ Vf, const float* __restrict__ freqs,
    uint16_t* __restrict__ Qb, uint16_t* __restrict__ Kb,
    uint16_t* __restrict__ Vb) {
  const int S = 2048, H = 16, G = 4;
  const size_t idx = (size_t)blockIdx.x * 256 + threadIdx.x;
  const size_t total = (size_t)4096 * 1536;
  if (idx >= total) return;
  const int row = (int)(idx / 1536);
  const int p   = (int)(idx % 1536);
  const int b = row >> 11;
  const int s = row & 2047;

  if (p < 1024) {  // Q
    const int h = p >> 6, dp = p & 63;
    const float q0 = Qf[(size_t)row * 2048 + h * 128 + 2 * dp];
    const float q1 = Qf[(size_t)row * 2048 + h * 128 + 2 * dp + 1];
    float sn, cs;
    __sincosf(freqs[s * 64 + dp], &sn, &cs);
    const size_t o = (((size_t)(b * H + h) * S + s) * 128) + 2 * dp;
    Qb[o]     = f16_bits(q0 * cs - q1 * sn);
    Qb[o + 1] = f16_bits(q0 * sn + q1 * cs);
  } else if (p < 1280) {  // K
    const int pp = p - 1024;
    const int g = pp >> 6, dp = pp & 63;
    const float k0 = Kf[(size_t)row * 512 + g * 128 + 2 * dp];
    const float k1 = Kf[(size_t)row * 512 + g * 128 + 2 * dp + 1];
    float sn, cs;
    __sincosf(freqs[s * 64 + dp], &sn, &cs);
    const size_t o = (((size_t)(b * G + g) * S + s) * 128) + 2 * dp;
    Kb[o]     = f16_bits(k0 * cs - k1 * sn);
    Kb[o + 1] = f16_bits(k0 * sn + k1 * cs);
  } else {  // V (no rope)
    const int pp = p - 1280;
    const int g = pp >> 6, dp = pp & 63;
    const size_t o = (((size_t)(b * G + g) * S + s) * 128) + 2 * dp;
    Vb[o]     = f16_bits(Vf[(size_t)row * 512 + g * 128 + 2 * dp]);
    Vb[o + 1] = f16_bits(Vf[(size_t)row * 512 + g * 128 + 2 * dp + 1]);
  }
}

// ---------------------------------------------------------------------------
// Kernel 3: flash attention. Block = 128 threads (4 waves); block handles one
// (b, h, 64-query tile); each wave owns 16 queries x full head_dim output.
// Per 32-key iteration: 8 WMMAs (Q@K^T over 4 d-chunks x 2 key subtiles),
// online softmax with shfl_xor row reductions, P via per-wave LDS transpose,
// 8 WMMAs (P@V over 8 d-subtiles). Next K/V tiles are prefetched
// (global_prefetch_b8) while the current tile's WMMAs run.
// ---------------------------------------------------------------------------
__global__ __launch_bounds__(128) void attn_wmma(
    const uint16_t* __restrict__ Qb, const uint16_t* __restrict__ Kb,
    const uint16_t* __restrict__ Vb, float* __restrict__ Ob) {
  const int S = 2048, H = 16, G = 4;
  __shared__ uint32_t Kl[32][65];      // [key][d-pair] f16x2
  __shared__ uint32_t Vl[128][17];     // [d][key-pair] f16x2 (transposed)
  __shared__ uint16_t Pl[4][16][34];   // per-wave P tile [q][key] f16

  const int tid  = threadIdx.x;
  const int lane = tid & 31;
  const int wave = tid >> 5;
  const int l16  = lane & 15;
  const int hl   = lane >> 4;
  const int h = blockIdx.y;
  const int b = blockIdx.z;
  const int g = h >> 2;  // GQA: repeat_interleave -> group = h / 4
  const int qbase = blockIdx.x * 64 + wave * 16;

  // Load Q A-fragments for this wave's 16 query rows (4 d-chunks of 32).
  const uint32_t* Qrow =
      (const uint32_t*)Qb + (((size_t)(b * H + h) * S + qbase + l16) * 128) / 2;
  Frag16 Aq[4];
#pragma unroll
  for (int c = 0; c < 4; ++c) {
    const int base = 16 * c + (hl ? 4 : 0);
#pragma unroll
    for (int i = 0; i < 4; ++i) {
      Aq[c].u[i]     = Qrow[base + i];
      Aq[c].u[4 + i] = Qrow[base + 8 + i];
    }
  }

  float run_m[8], run_s[8];
#pragma unroll
  for (int r = 0; r < 8; ++r) { run_m[r] = -1e30f; run_s[r] = 0.f; }
  v8f O[8];
#pragma unroll
  for (int nt = 0; nt < 8; ++nt)
    O[nt] = (v8f){0.f, 0.f, 0.f, 0.f, 0.f, 0.f, 0.f, 0.f};

  const uint32_t* Kg = (const uint32_t*)Kb + ((size_t)(b * G + g) * S) * 64;
  const uint16_t* Vg = Vb + ((size_t)(b * G + g) * S) * 128;
  const float scale = 0.08838834764831845f;  // 1/sqrt(128)

  for (int s0 = 0; s0 < S; s0 += 32) {
    __syncthreads();
    // Stage K tile: 32 keys x 64 d-pairs, direct u32 copy (coalesced in d).
#pragma unroll
    for (int i = 0; i < 16; ++i) {
      const int idx = tid + 128 * i;
      const int key = idx >> 6, dp = idx & 63;
      Kl[key][dp] = Kg[(size_t)(s0 + key) * 64 + dp];
    }
    // Stage V tile transposed: [d][key-pair] (coalesced in d).
#pragma unroll
    for (int i = 0; i < 16; ++i) {
      const int idx = tid + 128 * i;
      const int d = idx & 127, kp = idx >> 7;
      const uint32_t v0 = Vg[(size_t)(s0 + 2 * kp) * 128 + d];
      const uint32_t v1 = Vg[(size_t)(s0 + 2 * kp + 1) * 128 + d];
      Vl[d][kp] = v0 | (v1 << 16);
    }
    __syncthreads();
    // Prefetch next K/V tiles (8KB each; 128 threads x 64B stride).
    if (s0 + 32 < S) {
      __builtin_prefetch(Kg + (size_t)(s0 + 32) * 64 + tid * 16, 0, 0);
      __builtin_prefetch(Vg + (size_t)(s0 + 32) * 128 + tid * 32, 0, 0);
    }

    // Scores: S = Q @ K^T (16q x 32keys) as two 16x16 C tiles.
    v8f Sc0 = (v8f){0.f, 0.f, 0.f, 0.f, 0.f, 0.f, 0.f, 0.f};
    v8f Sc1 = Sc0;
#pragma unroll
    for (int c = 0; c < 4; ++c) {
      Frag16 B0, B1;
      const int base = 16 * c + (hl ? 4 : 0);
#pragma unroll
      for (int i = 0; i < 4; ++i) {
        B0.u[i]     = Kl[l16][base + i];       // keys 0..15 columns
        B0.u[4 + i] = Kl[l16][base + 8 + i];
        B1.u[i]     = Kl[16 + l16][base + i];  // keys 16..31 columns
        B1.u[4 + i] = Kl[16 + l16][base + 8 + i];
      }
      Sc0 = WMMA_F16(Aq[c].v, B0.v, Sc0);
      Sc1 = WMMA_F16(Aq[c].v, B1.v, Sc1);
    }

    // Online softmax. Row m = r + 8*hl lives across the 16-lane half.
    float al[8];
#pragma unroll
    for (int r = 0; r < 8; ++r) {
      const float a0 = Sc0[r] * scale;
      const float a1 = Sc1[r] * scale;
      float mx = fmaxf(a0, a1);
      mx = fmaxf(mx, __shfl_xor(mx, 1, 32));
      mx = fmaxf(mx, __shfl_xor(mx, 2, 32));
      mx = fmaxf(mx, __shfl_xor(mx, 4, 32));
      mx = fmaxf(mx, __shfl_xor(mx, 8, 32));
      const float nm  = fmaxf(run_m[r], mx);
      const float alr = __expf(run_m[r] - nm);
      run_m[r] = nm;
      const float p0 = __expf(a0 - nm);
      const float p1 = __expf(a1 - nm);
      float rs = p0 + p1;
      rs += __shfl_xor(rs, 1, 32);
      rs += __shfl_xor(rs, 2, 32);
      rs += __shfl_xor(rs, 4, 32);
      rs += __shfl_xor(rs, 8, 32);
      run_s[r] = run_s[r] * alr + rs;
      al[r] = alr;
      const int m = r + hl * 8;
      Pl[wave][m][l16]      = f16_bits(p0);
      Pl[wave][m][16 + l16] = f16_bits(p1);
    }
#pragma unroll
    for (int nt = 0; nt < 8; ++nt)
#pragma unroll
      for (int r = 0; r < 8; ++r) O[nt][r] *= al[r];

    // P as A-fragment (per-wave LDS; same-wave LDS ops stay in order).
    Frag16 Pf;
#pragma unroll
    for (int i = 0; i < 4; ++i) {
      const int j0 = (hl ? 4 : 0) + i;
      Pf.u[i]     = *(const uint32_t*)&Pl[wave][l16][2 * j0];
      Pf.u[4 + i] = *(const uint32_t*)&Pl[wave][l16][2 * (8 + j0)];
    }
    // O += P @ V over 8 d-subtiles.
#pragma unroll
    for (int nt = 0; nt < 8; ++nt) {
      Frag16 Bv;
      const int d = nt * 16 + l16;
      const int base = hl ? 4 : 0;
#pragma unroll
      for (int i = 0; i < 4; ++i) {
        Bv.u[i]     = Vl[d][base + i];
        Bv.u[4 + i] = Vl[d][base + 8 + i];
      }
      O[nt] = WMMA_F16(Pf.v, Bv.v, O[nt]);
    }
  }

  // Epilogue: write [B, S, H, 128] f32 so the projection sees row-major rows.
#pragma unroll
  for (int r = 0; r < 8; ++r) {
    const float inv = 1.f / run_s[r];
    const int sq = qbase + r + hl * 8;
#pragma unroll
    for (int nt = 0; nt < 8; ++nt) {
      const int d = nt * 16 + l16;
      Ob[(((size_t)b * S + sq) * H + h) * 128 + d] = O[nt][r] * inv;
    }
  }
}

// ---------------------------------------------------------------------------
// Launch. Workspace layout (bytes), ~75.5 MB total:
//   QF f32 [4096,2048] @ 0          (reused as attention output OB)
//   KF f32 [4096, 512] @ 33554432
//   VF f32 [4096, 512] @ 41943040
//   QB f16 [B,H,S,128] @ 50331648
//   KB f16 [B,G,S,128] @ 67108864
//   VB f16 [B,G,S,128] @ 71303168
// ---------------------------------------------------------------------------
extern "C" void kernel_launch(void* const* d_in, const int* in_sizes, int n_in,
                              void* d_out, int out_size, void* d_ws, size_t ws_size,
                              hipStream_t stream) {
  const float* x     = (const float*)d_in[0];
  const float* freqs = (const float*)d_in[1];
  const float* Wq    = (const float*)d_in[2];
  const float* bq    = (const float*)d_in[3];
  const float* Wk    = (const float*)d_in[4];
  const float* bk    = (const float*)d_in[5];
  const float* Wv    = (const float*)d_in[6];
  const float* bv    = (const float*)d_in[7];
  const float* Wo    = (const float*)d_in[8];
  const float* bo    = (const float*)d_in[9];
  float* out = (float*)d_out;

  char* ws = (char*)d_ws;
  float*    QF = (float*)(ws + 0);
  float*    KF = (float*)(ws + 33554432);
  float*    VF = (float*)(ws + 41943040);
  uint16_t* QB = (uint16_t*)(ws + 50331648);
  uint16_t* KB = (uint16_t*)(ws + 67108864);
  uint16_t* VB = (uint16_t*)(ws + 71303168);
  float*    OB = QF;  // QF dead after rope_pack; reuse for attention output.

  const int M = 4096, D = 2048, KVN = 512;

  // QKV projections.
  gemm_f16_wmma<<<dim3(D / 128, M / 128), 256, 0, stream>>>(x, Wq, bq, QF, M, D, D);
  gemm_f16_wmma<<<dim3(KVN / 128, M / 128), 256, 0, stream>>>(x, Wk, bk, KF, M, KVN, D);
  gemm_f16_wmma<<<dim3(KVN / 128, M / 128), 256, 0, stream>>>(x, Wv, bv, VF, M, KVN, D);

  // RoPE + f16 pack + transpose.
  const size_t total = (size_t)4096 * 1536;
  rope_pack<<<dim3((unsigned)((total + 255) / 256)), 256, 0, stream>>>(
      QF, KF, VF, freqs, QB, KB, VB);

  // Flash attention: grid (S/64 q-tiles, H, B).
  attn_wmma<<<dim3(2048 / 64, 16, 2), 128, 0, stream>>>(QB, KB, VB, OB);

  // Output projection into d_out.
  gemm_f16_wmma<<<dim3(D / 128, M / 128), 256, 0, stream>>>(OB, Wo, bo, out, M, D, D);
}
